// GraphFeat_24498493456643
// MI455X (gfx1250) — matrix operand order repaired and checked by reading, not compiled, not run on hardware
//
#include <hip/hip_runtime.h>
#include <hip/hip_bf16.h>

typedef __attribute__((ext_vector_type(16))) _Float16     v16h;
typedef __attribute__((ext_vector_type(8)))  float        v8f;
typedef __attribute__((ext_vector_type(4)))  unsigned int u32x4;
typedef __attribute__((ext_vector_type(4)))  float        f32x4;

union V16HQ { v16h v; u32x4 q[2]; _Float16 h[16]; };
union PK8   { _Float16 h[8]; u32x4 q; };

// ---------------------------------------------------------------------------
// K1: conv1x1 64->32 + bias + relu.  GEMM M=32(out ch), N=16 pixels, K=64.
// x: [B,64,32,64] f32 (NCHW).  h1 out: [B, HW, 32] f16 (HWC, ch contiguous).
// One wave = one 16x16 output tile (2 K-steps of 32).
// ---------------------------------------------------------------------------
__global__ __launch_bounds__(256) void k_conv1(const float* __restrict__ x,
                                               const float* __restrict__ w11,
                                               const float* __restrict__ b11,
                                               _Float16* __restrict__ h1) {
    const int lane  = threadIdx.x & 31;
    const int wave  = threadIdx.x >> 5;
    const int job   = blockIdx.x * 8 + wave;      // 131072 jobs
    const int mtile = job & 1;                    // 2 tiles of 16 out-ch
    const int ptile = job >> 1;                   // 65536 pixel tiles
    const int b     = ptile >> 7;                 // 128 tiles per image
    const int hw    = ((ptile & 127) << 4) + (lane & 15);
    const int half  = lane >> 4;
    const int lrow  = lane & 15;

    const float* xb   = x + (size_t)b * 64 * 2048 + hw;
    const float* wrow = w11 + (size_t)(mtile * 16 + lrow) * 64;

    v8f c = {};
    for (int kk = 0; kk < 2; ++kk) {
        const int kb = kk * 32;
        V16HQ a, bb;
#pragma unroll
        for (int e = 0; e < 16; ++e) {
            // A-matrix lane layout: K = half*16 + e
            int ka = kb + half * 16 + e;
            a.h[e] = (_Float16)wrow[ka];
            // B-matrix lane layout: K = half*16 + e (coalesced over hw lanes)
            int kv = kb + half * 16 + e;
            bb.h[e] = (_Float16)xb[(size_t)kv * 2048];
        }
        c = __builtin_amdgcn_wmma_f32_16x16x32_f16(false, a.v, false, bb.v,
                                                   (short)0, c, false, false);
    }
    PK8 out;
#pragma unroll
    for (int r = 0; r < 8; ++r) {
        float v = c[r] + b11[mtile * 16 + half * 8 + r];
        out.h[r] = (_Float16)(v > 0.f ? v : 0.f);
    }
    // lane's 8 channels are contiguous: [mtile*16 + half*8, +8)  -> one b128
    _Float16* dst = h1 + ((size_t)b * 2048 + hw) * 32 + mtile * 16 + half * 8;
    *(u32x4*)dst = out.q;
}

// ---------------------------------------------------------------------------
// K2: conv3x3 32->16 pad1 + bias + relu, implicit GEMM, LDS band tiling.
// Block = 1 image x 8-row band (+1-row halo).  9 K-steps of 32 (one per tap).
// h1 in: [B,HW,32] f16   h2 out: [B,HW,16] f16
// ---------------------------------------------------------------------------
#define K2_PS 40   // padded per-pixel channel stride (f16): 80B, 16B-aligned
__global__ __launch_bounds__(256) void k_conv2(const _Float16* __restrict__ h1,
                                               const float* __restrict__ w12,
                                               const float* __restrict__ b12,
                                               _Float16* __restrict__ h2) {
    __shared__ __align__(16) _Float16 ldsA[10 * 66 * K2_PS]; // [row][col][ch]
    __shared__ __align__(16) _Float16 ldsW[9 * 16 * K2_PS];  // [tap][m][ch]
    const int tid  = threadIdx.x;
    const int b    = blockIdx.x >> 2;
    const int band = blockIdx.x & 3;
    const int r0   = band * 8;

    // stage weights (w12 flat [16][32][3][3]: i == (m*32+c)*9 + t)
    for (int i = tid; i < 16 * 32 * 9; i += 256) {
        int t = i % 9, c = (i / 9) % 32, m = i / 288;
        ldsW[(t * 16 + m) * K2_PS + c] = (_Float16)w12[i];
    }
    // stage activation band with zero halo: 10 x 66 pixels, 64B each
    for (int p = tid; p < 10 * 66; p += 256) {
        int rr = p / 66, cc = p % 66;
        int gr = r0 + rr - 1, gc = cc - 1;
        u32x4 z0 = {0,0,0,0}, z1 = {0,0,0,0}, z2 = {0,0,0,0}, z3 = {0,0,0,0};
        if (gr >= 0 && gr < 32 && gc >= 0 && gc < 64) {
            const u32x4* src =
                (const u32x4*)(h1 + ((size_t)b * 2048 + gr * 64 + gc) * 32);
            z0 = src[0]; z1 = src[1]; z2 = src[2]; z3 = src[3];
        }
        u32x4* dst = (u32x4*)(ldsA + (size_t)p * K2_PS);
        dst[0] = z0; dst[1] = z1; dst[2] = z2; dst[3] = z3;
    }
    __syncthreads();

    const int lane = tid & 31, wave = tid >> 5;
    const int half = lane >> 4, lrow = lane & 15;
    for (int tile = wave; tile < 32; tile += 8) {
        const int p  = tile * 16 + lrow;          // pixel in band: 0..511
        const int rr = p >> 6, cc = p & 63;
        v8f acc = {};
#pragma unroll
        for (int t = 0; t < 9; ++t) {
            const int dy = t / 3, dx = t % 3;
            const u32x4* aw = (const u32x4*)(ldsW + (t * 16 + lrow) * K2_PS);
            V16HQ a;
            a.q[0] = aw[2 * half];                // K = half*16 + [0,8)
            a.q[1] = aw[2 * half + 1];            // K = half*16 + 8 + [0,8)
            const u32x4* bp = (const u32x4*)
                (ldsA + ((rr + dy) * 66 + (cc + dx)) * K2_PS + half * 16);
            V16HQ bv;
            bv.q[0] = bp[0]; bv.q[1] = bp[1];     // K = half*16 + [0,16)
            acc = __builtin_amdgcn_wmma_f32_16x16x32_f16(
                false, a.v, false, bv.v, (short)0, acc, false, false);
        }
        PK8 out;
#pragma unroll
        for (int r = 0; r < 8; ++r) {
            float v = acc[r] + b12[half * 8 + r];
            out.h[r] = (_Float16)(v > 0.f ? v : 0.f);
        }
        _Float16* dst =
            h2 + ((size_t)b * 2048 + (r0 + rr) * 64 + cc) * 16 + half * 8;
        *(u32x4*)dst = out.q;
    }
}

// ---------------------------------------------------------------------------
// K3: conv1x1 16->8 + relu (VALU: M=8 would waste a WMMA tile; BW-bound).
// h2: [B,HW,16] f16 -> h3: [B,HW,8] f16
// ---------------------------------------------------------------------------
__global__ __launch_bounds__(256) void k_conv3(const _Float16* __restrict__ h2,
                                               const float* __restrict__ w21,
                                               const float* __restrict__ b21,
                                               _Float16* __restrict__ h3) {
    const int pix = blockIdx.x * 256 + threadIdx.x;   // B*2048
    const _Float16* src = h2 + (size_t)pix * 16;
    float in[16];
#pragma unroll
    for (int c = 0; c < 16; ++c) in[c] = (float)src[c];
    PK8 out;
#pragma unroll
    for (int oc = 0; oc < 8; ++oc) {
        float acc = b21[oc];
#pragma unroll
        for (int c = 0; c < 16; ++c) acc += in[c] * w21[oc * 16 + c];
        out.h[oc] = (_Float16)(acc > 0.f ? acc : 0.f);
    }
    *(u32x4*)(h3 + (size_t)pix * 8) = out.q;
}

// ---------------------------------------------------------------------------
// K4: conv3x3 8->4 pad1 + relu (VALU).  h3: [B,HW,8] f16
// h4 out: [B,8192] f16 in CHW order == reference's reshape order for the FC.
// ---------------------------------------------------------------------------
__global__ __launch_bounds__(256) void k_conv4(const _Float16* __restrict__ h3,
                                               const float* __restrict__ w22,
                                               const float* __restrict__ b22,
                                               _Float16* __restrict__ h4) {
    const int pix = blockIdx.x * 256 + threadIdx.x;
    const int b = pix >> 11, hw = pix & 2047, hh = hw >> 6, ww = hw & 63;
    float a0 = b22[0], a1 = b22[1], a2 = b22[2], a3 = b22[3];
#pragma unroll
    for (int dy = 0; dy < 3; ++dy) {
        const int y = hh + dy - 1;
#pragma unroll
        for (int dx = 0; dx < 3; ++dx) {
            const int xx = ww + dx - 1;
            if (y < 0 || y >= 32 || xx < 0 || xx >= 64) continue;
            const _Float16* src = h3 + ((size_t)b * 2048 + y * 64 + xx) * 8;
#pragma unroll
            for (int c = 0; c < 8; ++c) {
                float v = (float)src[c];
                a0 += v * w22[((0 * 8 + c) * 3 + dy) * 3 + dx];
                a1 += v * w22[((1 * 8 + c) * 3 + dy) * 3 + dx];
                a2 += v * w22[((2 * 8 + c) * 3 + dy) * 3 + dx];
                a3 += v * w22[((3 * 8 + c) * 3 + dy) * 3 + dx];
            }
        }
    }
    _Float16* dst = h4 + (size_t)b * 8192 + hw;
    dst[0]    = (_Float16)fmaxf(a0, 0.f);
    dst[2048] = (_Float16)fmaxf(a1, 0.f);
    dst[4096] = (_Float16)fmaxf(a2, 0.f);
    dst[6144] = (_Float16)fmaxf(a3, 0.f);
}

// ---------------------------------------------------------------------------
// K5: FC GEMM [512 x 8192] * fcw^T [8192 x 256] + bias, sigmoid(3z)*scale.
// One wave per 16x16 (batch x out) tile; K-loop 256 steps of 32.
// fcw (8.4MB) + h4 (8.4MB) stay L2-resident (192MB L2).
// ---------------------------------------------------------------------------
__global__ __launch_bounds__(256) void k_fc(const _Float16* __restrict__ h4,
                                            const float* __restrict__ fcw,
                                            const float* __restrict__ fcb,
                                            float* __restrict__ coord_out) {
    const int lane  = threadIdx.x & 31, wave = threadIdx.x >> 5;
    const int job   = blockIdx.x * 8 + wave;       // 512 = 32 mtiles * 16 ntiles
    const int mtile = job >> 4, ntile = job & 15;
    const int half  = lane >> 4, lrow = lane & 15;
    const int batch = mtile * 16 + lrow;           // A row (M = lane%16)
    const int out_n = ntile * 16 + lrow;           // B col (N = lane%16)

    const _Float16* arow = h4 + (size_t)batch * 8192;
    const float*    brow = fcw + (size_t)out_n * 8192;

    v8f acc = {};
    for (int k0 = 0; k0 < 8192; k0 += 32) {
        __builtin_prefetch(brow + k0 + 1024, 0, 1);  // global_prefetch
        V16HQ a, bv;
        a.q[0] = *(const u32x4*)(arow + k0 + half * 16);
        a.q[1] = *(const u32x4*)(arow + k0 + half * 16 + 8);
        const f32x4* b4 = (const f32x4*)(brow + k0 + half * 16);
        f32x4 f0 = b4[0], f1 = b4[1], f2 = b4[2], f3 = b4[3];
#pragma unroll
        for (int e = 0; e < 4; ++e) {
            bv.h[e]      = (_Float16)f0[e];
            bv.h[4 + e]  = (_Float16)f1[e];
            bv.h[8 + e]  = (_Float16)f2[e];
            bv.h[12 + e] = (_Float16)f3[e];
        }
        acc = __builtin_amdgcn_wmma_f32_16x16x32_f16(false, a.v, false, bv.v,
                                                     (short)0, acc, false, false);
    }
    const float bias  = fcb[out_n];
    const float scale = (out_n & 1) ? 63.0f : 31.0f;   // (W-1, H-1) interleave
#pragma unroll
    for (int r = 0; r < 8; ++r) {
        const int bm = mtile * 16 + half * 8 + r;      // D row M = r + 8*half
        float z = acc[r] + bias;
        float s = 1.0f / (1.0f + __expf(-3.0f * z));
        coord_out[(size_t)bm * 256 + out_n] = s * scale;
    }
}

// ---------------------------------------------------------------------------
// K6: bilinear gather.  thread = (b, node, c).  Matches reference corner order.
// ---------------------------------------------------------------------------
__global__ __launch_bounds__(256) void k_gather(const float* __restrict__ x,
                                                const float* __restrict__ coord,
                                                float* __restrict__ gf) {
    const int t = blockIdx.x * 256 + threadIdx.x;   // 512*128*64
    const int c = t & 63, node = (t >> 6) & 127, b = t >> 13;
    float ch = coord[((size_t)b * 128 + node) * 2 + 0];
    float cw = coord[((size_t)b * 128 + node) * 2 + 1];
    ch = fminf(fmaxf(ch, 0.f), 31.f);
    cw = fminf(fmaxf(cw, 0.f), 63.f);
    const int h0 = (int)floorf(ch), h1i = (int)ceilf(ch);
    const int w0 = (int)floorf(cw), w1i = (int)ceilf(cw);
    const float oh = ch - (float)h0, ow = cw - (float)w0;
    const float* xc = x + ((size_t)b * 64 + c) * 2048;
    const float v_lt = xc[h0 * 64 + w0];
    const float v_rt = xc[h1i * 64 + w0];
    const float v_lb = xc[h0 * 64 + w1i];
    const float v_rb = xc[h1i * 64 + w1i];
    const float vt = v_lt + oh * (v_rt - v_lt);
    const float vb = v_lb + oh * (v_rb - v_lb);
    gf[t] = vt + ow * (vb - vt);
}

// ---------------------------------------------------------------------------
extern "C" void kernel_launch(void* const* d_in, const int* in_sizes, int n_in,
                              void* d_out, int out_size, void* d_ws, size_t ws_size,
                              hipStream_t stream) {
    const float* x   = (const float*)d_in[0];
    const float* w11 = (const float*)d_in[1];
    const float* b11 = (const float*)d_in[2];
    const float* w12 = (const float*)d_in[3];
    const float* b12 = (const float*)d_in[4];
    const float* w21 = (const float*)d_in[5];
    const float* b21 = (const float*)d_in[6];
    const float* w22 = (const float*)d_in[7];
    const float* b22 = (const float*)d_in[8];
    const float* fcw = (const float*)d_in[9];
    const float* fcb = (const float*)d_in[10];

    float* out   = (float*)d_out;
    float* gf    = out;                 // graph_feat: 512*128*64
    float* coord = out + 4194304;       // coord:      512*128*2

    char* ws = (char*)d_ws;
    _Float16* h1 = (_Float16*)(ws);                  //  67,108,864 B [B,HW,32]
    _Float16* h2 = (_Float16*)(ws +  67108864);      //  33,554,432 B [B,HW,16]
    _Float16* h3 = (_Float16*)(ws + 100663296);      //  16,777,216 B [B,HW,8]
    _Float16* h4 = (_Float16*)(ws + 117440512);      //   8,388,608 B [B,8192]

    k_conv1 <<<16384, 256, 0, stream>>>(x,  w11, b11, h1);
    k_conv2 <<< 2048, 256, 0, stream>>>(h1, w12, b12, h2);
    k_conv3 <<< 4096, 256, 0, stream>>>(h2, w21, b21, h3);
    k_conv4 <<< 4096, 256, 0, stream>>>(h3, w22, b22, h4);
    k_fc    <<<   64, 256, 0, stream>>>(h4, fcw, fcb, coord);
    k_gather<<<16384, 256, 0, stream>>>(x,  coord, gf);
}
